// Head_15839839387892
// MI455X (gfx1250) — matrix operand order repaired and checked by reading, not compiled, not run on hardware
//
#include <hip/hip_runtime.h>
#include <hip/hip_bf16.h>

// ---------------------------------------------------------------------------
// Causal attention head, B=4 T=4096 C=1024 H=64, flash-attention style,
// f16 WMMA (v_wmma_f32_16x16x32_f16) with f32 accumulation. Wave32.
// Round 4: amdgpu_waves_per_eu(1) to lift the 128-VGPR cap, plus
// __builtin_amdgcn_sched_barrier(0) fences so the machine scheduler cannot
// sink the double-buffered B-tile loads back to their uses (rounds 2/3 kept
// re-serializing into one register pair with s_wait_loadcnt 0x0 per WMMA).
// ---------------------------------------------------------------------------

typedef __attribute__((ext_vector_type(16))) _Float16 v16h;
typedef __attribute__((ext_vector_type(8)))  _Float16 v8h;
typedef __attribute__((ext_vector_type(8)))  float    v8f;

#define Bn 4
#define Tn 4096
#define Cn 1024
#define Hn 64

#if defined(__has_builtin) && __has_builtin(__builtin_amdgcn_sched_barrier)
#define SCHED_FENCE() __builtin_amdgcn_sched_barrier(0)
#else
#define SCHED_FENCE() asm volatile("" ::: "memory")
#endif

__device__ __forceinline__ v8f wmma_f16(v16h a, v16h b, v8f c) {
  // D = A(16x32 f16) * B(32x16 f16) + C(16x16 f32)
  return __builtin_amdgcn_wmma_f32_16x16x32_f16(false, a, false, b,
                                                (short)0, c, false, false);
}

// Build a v16h A-operand from two contiguous 16B half-runs.
__device__ __forceinline__ v16h ld_a16(const _Float16* p0, const _Float16* p1) {
  v8h lo = *(const v8h*)p0;
  v8h hi = *(const v8h*)p1;
  v16h r;
#pragma unroll
  for (int i = 0; i < 8; ++i) { r[i] = lo[i]; r[i + 8] = hi[i]; }
  return r;
}

// ---------------------------------------------------------------------------
// Kernel 1: swizzle Wq/Wk/Wv (f32 [C,H]) into per-lane WMMA B-matrix layout,
// f16.  One wave per (w, ntile, ktile) B-tile (3*4*32 = 384 tiles).
// ---------------------------------------------------------------------------
__global__ void prep_w_kernel(const float* __restrict__ Wq,
                              const float* __restrict__ Wk,
                              const float* __restrict__ Wv,
                              _Float16* __restrict__ wswz) {
  const int gid  = blockIdx.x * blockDim.x + threadIdx.x;
  const int wave = gid >> 5;
  const int lane = gid & 31;
  if (wave >= 3 * 4 * 32) return;
  const int w  = wave / 128;          // 0=q 1=k 2=v
  const int nt = (wave % 128) / 32;   // output column tile
  const int kt = wave % 32;           // K tile
  const float* W = (w == 0) ? Wq : (w == 1) ? Wk : Wv;
  const int n  = nt * 16 + (lane & 15);
  const int c0 = kt * 32 + ((lane < 16) ? 0 : 8);
  _Float16* dst = wswz + ((size_t)wave * 32 + lane) * 16;
#pragma unroll
  for (int j = 0; j < 16; ++j)
    dst[j] = (_Float16)W[(size_t)(c0 + j + (j >= 8 ? 8 : 0)) * Hn + n];
}

// ---------------------------------------------------------------------------
// Kernel 2: QKV projection.  One wave computes a 16-row tile of q,k,v.
// 12 WMMAs per 32-wide K step; B tiles double-buffered across K steps.
// ---------------------------------------------------------------------------
__device__ __forceinline__ v16h ld_xa(const float* xrow, int kt, int lh) {
  // A layout (16x32 f16): lane l row = l&15; halves 0..7 = K lh*8+0..7,
  // halves 8..15 = K 16+lh*8+0..7.
  const int ka = kt * 32 + lh * 8;
  const float4 f0 = *(const float4*)(xrow + ka);
  const float4 f1 = *(const float4*)(xrow + ka + 4);
  const float4 f2 = *(const float4*)(xrow + ka + 16);
  const float4 f3 = *(const float4*)(xrow + ka + 20);
  v16h a;
  a[0]  = (_Float16)f0.x; a[1]  = (_Float16)f0.y;
  a[2]  = (_Float16)f0.z; a[3]  = (_Float16)f0.w;
  a[4]  = (_Float16)f1.x; a[5]  = (_Float16)f1.y;
  a[6]  = (_Float16)f1.z; a[7]  = (_Float16)f1.w;
  a[8]  = (_Float16)f2.x; a[9]  = (_Float16)f2.y;
  a[10] = (_Float16)f2.z; a[11] = (_Float16)f2.w;
  a[12] = (_Float16)f3.x; a[13] = (_Float16)f3.y;
  a[14] = (_Float16)f3.z; a[15] = (_Float16)f3.w;
  return a;
}

__device__ __forceinline__ void ld_btiles(const _Float16* __restrict__ wswz,
                                          int kt, int lane, v16h bt[12]) {
#pragma unroll
  for (int i = 0; i < 12; ++i)
    bt[i] = *(const v16h*)(wswz + ((size_t)(i * 32 + kt) * 32 + lane) * 16);
}

__global__ void __launch_bounds__(128)
__attribute__((amdgpu_waves_per_eu(1)))
qkv_proj_kernel(const float* __restrict__ x, const _Float16* __restrict__ wswz,
                _Float16* __restrict__ qh, _Float16* __restrict__ kh,
                _Float16* __restrict__ vt) {
  const int lane = threadIdx.x & 31;
  const int ln = lane & 15;
  const int lh = lane >> 4;
  const int tile = blockIdx.x * (blockDim.x >> 5) + (threadIdx.x >> 5);
  const int row0 = tile * 16;
  const float* xrow = x + (size_t)(row0 + ln) * Cn;

  v8f zero = {};
  v8f acc[12];
#pragma unroll
  for (int i = 0; i < 12; ++i) acc[i] = zero;

  v16h a0 = ld_xa(xrow, 0, lh);
  v16h b0[12];
  ld_btiles(wswz, 0, lane, b0);

  for (int kt = 0; kt < Cn / 32; kt += 2) {
    // Stage kt+1; fence so these loads stay hoisted above the kt WMMAs.
    v16h a1 = ld_xa(xrow, kt + 1, lh);
    v16h b1[12];
    ld_btiles(wswz, kt + 1, lane, b1);
    SCHED_FENCE();
#pragma unroll
    for (int i = 0; i < 12; ++i) acc[i] = wmma_f16(a0, b0[i], acc[i]);
    SCHED_FENCE();

    // Stage kt+2 (clamped on last trip); compute kt+1.
    const int kt2 = (kt + 2 < Cn / 32) ? kt + 2 : (Cn / 32 - 1);
    a0 = ld_xa(xrow, kt2, lh);
    ld_btiles(wswz, kt2, lane, b0);
    SCHED_FENCE();
#pragma unroll
    for (int i = 0; i < 12; ++i) acc[i] = wmma_f16(a1, b1[i], acc[i]);
    SCHED_FENCE();
  }

  // D layout: VGPR r, lane l -> row (r + 8*lh), col (l&15).
  const int bidx = row0 / Tn;
  const int tloc = row0 % Tn;
#pragma unroll
  for (int nt = 0; nt < 4; ++nt) {
    const int ch = nt * 16 + ln;
#pragma unroll
    for (int r = 0; r < 8; ++r) {
      const size_t t = (size_t)row0 + r + 8 * lh;
      qh[t * Hn + ch] = (_Float16)acc[0 * 4 + nt][r];
      kh[t * Hn + ch] = (_Float16)acc[1 * 4 + nt][r];
    }
    v8h pk;                                  // v transposed: 8 consecutive t
#pragma unroll
    for (int r = 0; r < 8; ++r) pk[r] = (_Float16)acc[2 * 4 + nt][r];
    *(v8h*)(vt + ((size_t)bidx * Hn + ch) * Tn + tloc + 8 * lh) = pk;
  }
}

// ---------------------------------------------------------------------------
// Kernel 3: flash attention.  One wave owns 16 query rows; iterates 32-key
// tiles with online softmax.  4 S-WMMAs + 4 O-WMMAs per tile.  All 8 B-tile
// loads are fenced above the compute so the softmax VALU work hides them.
// ---------------------------------------------------------------------------
__global__ void __launch_bounds__(128)
__attribute__((amdgpu_waves_per_eu(1)))
attn_kernel(const _Float16* __restrict__ qh, const _Float16* __restrict__ kh,
            const _Float16* __restrict__ vt, float* __restrict__ out) {
  __shared__ __align__(16) _Float16 pbuf[4][16][32];  // wave-private P staging
  const int lane = threadIdx.x & 31;
  const int ln = lane & 15;
  const int lh = lane >> 4;
  const int wv = threadIdx.x >> 5;
  const int b  = blockIdx.y;
  const int qbase = (blockIdx.x * 4 + wv) * 16;

  const _Float16* qp = qh + ((size_t)b * Tn + qbase) * Hn;
  const _Float16* kp = kh + (size_t)b * Tn * Hn;
  const _Float16* vp = vt + (size_t)b * Hn * Tn;

  // Q as two A tiles (channels 0..31, 32..63), loaded once.
  const _Float16* qrow = qp + (size_t)ln * Hn;
  const v16h qa0 = ld_a16(qrow + lh * 16,      qrow + lh * 16 + 8);
  const v16h qa1 = ld_a16(qrow + 32 + lh * 16, qrow + 32 + lh * 16 + 8);

  v8f zero = {};
  v8f o[4];
#pragma unroll
  for (int i = 0; i < 4; ++i) o[i] = zero;
  float mrow[8], lrow[8];
#pragma unroll
  for (int r = 0; r < 8; ++r) { mrow[r] = -1e30f; lrow[r] = 0.0f; }

  const float scale = 0.03125f;  // 1/sqrt(C) = 1/32
  const int ntiles = (qbase + 16 + 31) / 32;

  for (int jt = 0; jt < ntiles; ++jt) {
    const int jk = jt * 32;
    // K B-tiles: lane l -> key jk+(l&15) (+16 for second), ch lh*16+0..15.
    const _Float16* k0p = kp + (size_t)(jk + ln) * Hn + lh * 16;
    const v16h bk0a = *(const v16h*)(k0p);
    const v16h bk0b = *(const v16h*)(k0p + 32);
    const v16h bk1a = *(const v16h*)(k0p + 16 * Hn);
    const v16h bk1b = *(const v16h*)(k0p + 16 * Hn + 32);

    // V B-tiles (independent of softmax) issued early: lane l -> ch
    // nt*16+(l&15), keys jk+lh*16+0..15 contiguous in transposed V.
    v16h bv[4];
#pragma unroll
    for (int nt = 0; nt < 4; ++nt)
      bv[nt] = *(const v16h*)(vp + (size_t)(nt * 16 + ln) * Tn + jk + lh * 16);

    // Prefetch next key tile into WGP$ (gfx1250 global_prefetch_b8).
    if (jt + 1 < ntiles) {
      __builtin_prefetch(kp + (size_t)(jk + 32 + lane) * Hn, 0, 3);
      __builtin_prefetch(vp + (size_t)lane * Tn + jk + 32, 0, 3);
      __builtin_prefetch(vp + (size_t)(32 + lane) * Tn + jk + 32, 0, 3);
    }
    SCHED_FENCE();  // keep all loads above the compute pipeline

    v8f s0 = zero, s1 = zero;
    s0 = wmma_f16(qa0, bk0a, s0);
    s0 = wmma_f16(qa1, bk0b, s0);
    s1 = wmma_f16(qa0, bk1a, s1);
    s1 = wmma_f16(qa1, bk1b, s1);

    float asc[8];
#pragma unroll
    for (int r = 0; r < 8; ++r) {
      const int qi = qbase + r + 8 * lh;
      float e0 = (jk + ln      <= qi) ? s0[r] * scale : -1e30f;
      float e1 = (jk + 16 + ln <= qi) ? s1[r] * scale : -1e30f;
      float mx = fmaxf(e0, e1);
#pragma unroll
      for (int d = 1; d < 16; d <<= 1) mx = fmaxf(mx, __shfl_xor(mx, d, 32));
      const float mn = fmaxf(mrow[r], mx);
      const float p0 = __expf(e0 - mn);
      const float p1 = __expf(e1 - mn);
      float ps = p0 + p1;
#pragma unroll
      for (int d = 1; d < 16; d <<= 1) ps += __shfl_xor(ps, d, 32);
      const float al = __expf(mrow[r] - mn);
      lrow[r] = lrow[r] * al + ps;
      mrow[r] = mn;
      asc[r] = al;
      // P in row-major LDS: row (r+8*lh), cols ln and 16+ln.
      pbuf[wv][r + 8 * lh][ln]      = (_Float16)p0;
      pbuf[wv][r + 8 * lh][16 + ln] = (_Float16)p1;
    }
#pragma unroll
    for (int nt = 0; nt < 4; ++nt)
#pragma unroll
      for (int r = 0; r < 8; ++r) o[nt][r] *= asc[r];

    // DS ops are in-order within a wave; fence so the A-layout re-read sees
    // all 32 lanes' stores and the compiler keeps program order.
    asm volatile("s_wait_dscnt 0" ::: "memory");

    // Re-read P as A tile (16 rows x 32 keys).
    const _Float16* prow = &pbuf[wv][ln][lh * 16];
    const v16h pa = ld_a16(prow, prow + 8);

#pragma unroll
    for (int nt = 0; nt < 4; ++nt)
      o[nt] = wmma_f16(pa, bv[nt], o[nt]);
  }

  float* orow = out + ((size_t)b * Tn + qbase) * Hn;
#pragma unroll
  for (int nt = 0; nt < 4; ++nt) {
    const int ch = nt * 16 + ln;
#pragma unroll
    for (int r = 0; r < 8; ++r)
      orow[(size_t)(r + 8 * lh) * Hn + ch] = o[nt][r] / lrow[r];
  }
}

// ---------------------------------------------------------------------------
extern "C" void kernel_launch(void* const* d_in, const int* in_sizes, int n_in,
                              void* d_out, int out_size, void* d_ws,
                              size_t ws_size, hipStream_t stream) {
  const float* x  = (const float*)d_in[0];
  const float* Wq = (const float*)d_in[1];
  const float* Wk = (const float*)d_in[2];
  const float* Wv = (const float*)d_in[3];
  float* out = (float*)d_out;

  // Workspace: swizzled weights (384 KiB) + qh/kh (2 MiB each) + vt (2 MiB).
  _Float16* wswz = (_Float16*)d_ws;
  const size_t wswz_halves = (size_t)3 * 4 * 32 * 32 * 16;  // 196608
  _Float16* qh = wswz + wswz_halves;
  _Float16* kh = qh + (size_t)Bn * Tn * Hn;
  _Float16* vt = kh + (size_t)Bn * Tn * Hn;

  prep_w_kernel<<<96, 128, 0, stream>>>(Wq, Wk, Wv, wswz);
  qkv_proj_kernel<<<(Bn * Tn / 16) / 4, 128, 0, stream>>>(x, wswz, qh, kh, vt);
  attn_kernel<<<dim3(Tn / 64, Bn), 128, 0, stream>>>(qh, kh, vt, out);
}
